// LampGraphContext5_56367150792746
// MI455X (gfx1250) — matrix operand order repaired and checked by reading, not compiled, not run on hardware
//
#include <hip/hip_runtime.h>

typedef __attribute__((ext_vector_type(16))) _Float16 v16h;
typedef __attribute__((ext_vector_type(8)))  float    v8f;

#define FD 128   // feature dim

// ---------------- utility kernels ----------------

__global__ void zero_f32_kernel(float* __restrict__ p, long n4) {
    long i = (long)blockIdx.x * blockDim.x + threadIdx.x;
    if (i < n4) ((float4*)p)[i] = make_float4(0.f, 0.f, 0.f, 0.f);
}

__global__ void f32_to_f16_kernel(const float* __restrict__ in,
                                  _Float16* __restrict__ out, long n4) {
    long i = (long)blockIdx.x * blockDim.x + threadIdx.x;
    if (i >= n4) return;
    float4 v = ((const float4*)in)[i];
    _Float16 o[4] = { (_Float16)v.x, (_Float16)v.y, (_Float16)v.z, (_Float16)v.w };
    *(float2*)(out + i * 4) = *(float2*)o;
}

// Wt[n][k] = W[k][n], converted to f16 (so B-columns become contiguous rows)
__global__ void transpose_w_f16_kernel(const float* __restrict__ w,
                                       _Float16* __restrict__ wt) {
    int t = blockIdx.x * blockDim.x + threadIdx.x;   // 16384 threads
    int k = t >> 7, n = t & 127;
    wt[n * FD + k] = (_Float16)w[k * FD + n];
}

// deg[dst] += 1 per edge
__global__ void count_deg_kernel(const int* __restrict__ dst,
                                 float* __restrict__ deg, int nedges) {
    int e = blockIdx.x * blockDim.x + threadIdx.x;
    if (e < nedges) atomicAdd(&deg[dst[e]], 1.0f);
}

// in-place deg -> rsqrt(deg + 1)  (the +1 is the self loop)
__global__ void dinv_kernel(float* __restrict__ deg, int n) {
    int i = blockIdx.x * blockDim.x + threadIdx.x;
    if (i < n) deg[i] = rsqrtf(deg[i] + 1.0f);
}

// ---------------- WMMA GEMM: h[n,128] = xh[n,128] @ W[128,128] ----------------
// xh: f16 row-major; wt: f16 W-transposed (wt[n][k] = W[k][n]); h: f32 row-major.
// Block = 256 threads (8 waves); block covers 128 rows; wave covers 16 rows x 128 cols.
__global__ __launch_bounds__(256) void gemm_wmma_kernel(
        const _Float16* __restrict__ xh, const _Float16* __restrict__ wt,
        float* __restrict__ h, int nrows) {
    __shared__ _Float16 sW[FD * FD];                 // 32 KB: whole W^T tile
    {
        const float4* src = (const float4*)wt;       // 2048 x 16B chunks
        float4* dst = (float4*)sW;
        for (int i = threadIdx.x; i < (FD * FD) / 8; i += 256) dst[i] = src[i];
    }
    __syncthreads();

    const int wave = threadIdx.x >> 5;
    const int lane = threadIdx.x & 31;
    const int rowBase = blockIdx.x * 128 + wave * 16;
    if (rowBase >= nrows) return;

    const int l15   = lane & 15;                     // A-row / B-col / D-col within tile
    const int khalf = lane >> 4;                     // 0 or 1

    v8f acc[8] = {};                                 // 16x16 f32 tile per column-tile

    int arow = rowBase + l15;
    if (arow >= nrows) arow = nrows - 1;             // safe clamp (stores masked below)
    const _Float16* aptr = xh + (size_t)arow * FD;

#pragma unroll
    for (int ks = 0; ks < 4; ks++) {
        // A fragment: contiguous k order (same as B below).
        // lane<16: K = ks*32 + {0..15}; lane>=16: K = ks*32 + {16..31}
        const int ka = ks * 32 + khalf * 16;
        v16h a;
        ((float4*)&a)[0] = *(const float4*)(aptr + ka);
        ((float4*)&a)[1] = *(const float4*)(aptr + ka + 8);

        // B fragments: lanes 0-15 hold K = ks*32+0..15 of column l15,
        //              lanes 16-31 hold K = ks*32+16..31 (contiguous 16 halves).
        const int kb = ks * 32 + khalf * 16;
#pragma unroll
        for (int nt = 0; nt < 8; nt++) {
            v16h b;
            const _Float16* bptr = sW + (nt * 16 + l15) * FD + kb;
            ((float4*)&b)[0] = ((const float4*)bptr)[0];
            ((float4*)&b)[1] = ((const float4*)bptr)[1];
            acc[nt] = __builtin_amdgcn_wmma_f32_16x16x32_f16(
                false, a, false, b, (short)0, acc[nt], false, false);
        }
    }

    // D layout: VGPR r, lane L -> row = r + 8*(L>=16), col = L&15
#pragma unroll
    for (int nt = 0; nt < 8; nt++) {
        float* outp = h + (size_t)(rowBase + khalf * 8) * FD + nt * 16 + l15;
#pragma unroll
        for (int r = 0; r < 8; r++) {
            int row = rowBase + khalf * 8 + r;
            if (row < nrows) outp[(size_t)r * FD] = acc[nt][r];
        }
    }
}

// ---------------- edge scatter: agg[dst] += h[src] * dinv[src]*dinv[dst] ----------------
// 32 lanes per edge, 4 floats per lane.
__global__ void edge_agg_kernel(const int* __restrict__ src, const int* __restrict__ dst,
                                const float* __restrict__ dinv, const float* __restrict__ h,
                                float* __restrict__ agg, int nedges) {
    long t = (long)blockIdx.x * blockDim.x + threadIdx.x;
    int e = (int)(t >> 5);
    if (e >= nedges) return;
    int lane = (int)(t & 31);
    int s = src[e], d = dst[e];
    float norm = dinv[s] * dinv[d];
    float4 hv = *(const float4*)(h + (size_t)s * FD + lane * 4);
    float* outp = agg + (size_t)d * FD + lane * 4;
    atomicAdd(outp + 0, hv.x * norm);
    atomicAdd(outp + 1, hv.y * norm);
    atomicAdd(outp + 2, hv.z * norm);
    atomicAdd(outp + 3, hv.w * norm);
}

// ---------------- pool: g[d] += sum_n relu(agg[n,d] + h[n,d]*dinv[n]^2 + b[d]) ----------
// (self-loop term folded in here, avoiding a second atomic pass)
__global__ __launch_bounds__(128) void pool_relu_sum_kernel(
        const float* __restrict__ agg, const float* __restrict__ h,
        const float* __restrict__ dinv, const float* __restrict__ b,
        float* __restrict__ g, int nnodes) {
    int d = threadIdx.x;
    int n0 = blockIdx.x * 64;
    int nend = n0 + 64; if (nend > nnodes) nend = nnodes;
    float bias = b[d];
    float acc = 0.f;
    for (int n = n0; n < nend; n++) {
        float di = dinv[n];
        float v = agg[(size_t)n * FD + d] + h[(size_t)n * FD + d] * di * di + bias;
        acc += fmaxf(v, 0.f);
    }
    atomicAdd(&g[d], acc);
}

// ---------------- attention + FiLM head (one block, 128 threads) ----------------
__global__ __launch_bounds__(128) void head_kernel(
        const float* __restrict__ gbuf,                      // [3*128]
        const float* __restrict__ fc1_w, const float* __restrict__ fc1_b,
        const float* __restrict__ fc2_w, const float* __restrict__ fc2_b,
        const float* __restrict__ fc3_w, const float* __restrict__ fc3_b,
        const float* __restrict__ f1_w,  const float* __restrict__ f1_b,
        const float* __restrict__ f2_w,  const float* __restrict__ f2_b,
        float* __restrict__ out) {
    __shared__ float s_xin[3 * FD], s_h[FD], s_h2[FD], s_a[3];
    int i = threadIdx.x;
    s_xin[i] = gbuf[i]; s_xin[i + FD] = gbuf[i + FD]; s_xin[i + 2 * FD] = gbuf[i + 2 * FD];
    __syncthreads();

    float acc = fc1_b[i];
    for (int j = 0; j < 3 * FD; j++) acc += s_xin[j] * fc1_w[i * (3 * FD) + j];
    s_h[i] = fmaxf(acc, 0.f);
    __syncthreads();

    acc = fc2_b[i];
    for (int j = 0; j < FD; j++) acc += s_h[j] * fc2_w[i * FD + j];
    s_h2[i] = fmaxf(acc, 0.f);                       // relu before fc3
    __syncthreads();

    if (i < 3) {
        float l = fc3_b[i];
        for (int j = 0; j < FD; j++) l += s_h2[j] * fc3_w[i * FD + j];
        s_a[i] = l;
    }
    __syncthreads();
    if (i == 0) {
        float m = fmaxf(s_a[0], fmaxf(s_a[1], s_a[2]));
        float e0 = __expf(s_a[0] - m), e1 = __expf(s_a[1] - m), e2 = __expf(s_a[2] - m);
        float s = e0 + e1 + e2;
        s_a[0] = e0 / s; s_a[1] = e1 / s; s_a[2] = e2 / s;
    }
    __syncthreads();

    float xd = s_a[0] * s_xin[i] + s_a[1] * s_xin[FD + i] + s_a[2] * s_xin[2 * FD + i];
    s_h[i] = xd;                                     // reuse as x[128]
    __syncthreads();

    float ga = f1_b[i], be = f2_b[i];
    for (int j = 0; j < FD; j++) {
        float xj = s_h[j];
        ga += xj * f1_w[i * FD + j];
        be += xj * f2_w[i * FD + j];
    }
    out[i]      = tanhf(ga);
    out[FD + i] = tanhf(be);
}

// ---------------- launcher ----------------

extern "C" void kernel_launch(void* const* d_in, const int* in_sizes, int n_in,
                              void* d_out, int out_size, void* d_ws, size_t ws_size,
                              hipStream_t stream) {
    const float* X[3]  = { (const float*)d_in[0], (const float*)d_in[1], (const float*)d_in[2] };
    const int*   EI[3] = { (const int*)d_in[3], (const int*)d_in[4], (const int*)d_in[5] };
    const float* CW[3] = { (const float*)d_in[6], (const float*)d_in[8], (const float*)d_in[10] };
    const float* CB[3] = { (const float*)d_in[7], (const float*)d_in[9], (const float*)d_in[11] };

    const long nn = in_sizes[0] / FD;     // 50000 nodes
    const int  ne = in_sizes[3] / 2;      // 800000 edges

    // workspace carve-up (256B aligned)
    char* ws = (char*)d_ws;
    size_t off = 0;
    auto carve = [&](size_t bytes) { size_t o = off; off += (bytes + 255) & ~(size_t)255; return o; };
    _Float16* xh   = (_Float16*)(ws + carve((size_t)nn * FD * 2));
    _Float16* wt   = (_Float16*)(ws + carve((size_t)FD * FD * 2));
    float*    h    = (float*)   (ws + carve((size_t)nn * FD * 4));
    float*    agg  = (float*)   (ws + carve((size_t)nn * FD * 4));
    float*    dinv = (float*)   (ws + carve((size_t)nn * 4));
    float*    gbuf = (float*)   (ws + carve((size_t)3 * FD * 4));
    (void)ws_size;

    zero_f32_kernel<<<1, 96, 0, stream>>>(gbuf, 96);   // 384 floats

    const long featN  = nn * FD;                       // 6.4M
    const long feat4  = featN / 4;
    const long node4  = nn / 4;
    const int  nGemm  = (int)((nn + 127) / 128);
    const long eThr   = (long)ne * 32;

    for (int g = 0; g < 3; g++) {
        zero_f32_kernel<<<(int)((node4 + 255) / 256), 256, 0, stream>>>(dinv, node4);
        zero_f32_kernel<<<(int)((feat4 + 255) / 256), 256, 0, stream>>>(agg, feat4);

        count_deg_kernel<<<(ne + 255) / 256, 256, 0, stream>>>(EI[g] + ne, dinv, ne);
        dinv_kernel<<<(int)((nn + 255) / 256), 256, 0, stream>>>(dinv, (int)nn);

        f32_to_f16_kernel<<<(int)((feat4 + 255) / 256), 256, 0, stream>>>(X[g], xh, feat4);
        transpose_w_f16_kernel<<<(FD * FD) / 256, 256, 0, stream>>>(CW[g], wt);

        gemm_wmma_kernel<<<nGemm, 256, 0, stream>>>(xh, wt, h, (int)nn);

        edge_agg_kernel<<<(int)((eThr + 255) / 256), 256, 0, stream>>>(
            EI[g], EI[g] + ne, dinv, h, agg, ne);

        pool_relu_sum_kernel<<<(int)((nn + 63) / 64), 128, 0, stream>>>(
            agg, h, dinv, CB[g], gbuf + g * FD, (int)nn);
    }

    head_kernel<<<1, 128, 0, stream>>>(gbuf,
        (const float*)d_in[12], (const float*)d_in[13],
        (const float*)d_in[14], (const float*)d_in[15],
        (const float*)d_in[16], (const float*)d_in[17],
        (const float*)d_in[18], (const float*)d_in[19],
        (const float*)d_in[20], (const float*)d_in[21],
        (float*)d_out);
}
